// ViTMoE_77352361001112
// MI455X (gfx1250) — matrix operand order repaired and checked by reading, not compiled, not run on hardware
//
#include <hip/hip_runtime.h>
#include <math.h>

// ---------------------------------------------------------------------------
// Types for CDNA5 WMMA (wave32): bf16 A/B fragments (16 elems), f32 accum (8)
// ---------------------------------------------------------------------------
typedef __attribute__((ext_vector_type(16))) __bf16          v16bf;
typedef __attribute__((ext_vector_type(8)))  float           v8f;
typedef __attribute__((ext_vector_type(16))) unsigned short  us16;
typedef __attribute__((ext_vector_type(8)))  unsigned short  us8;

union FragU { us16 u; v16bf v; };

__device__ __forceinline__ unsigned short f2bf(float f) {
  unsigned int x = __float_as_uint(f);
  x += 0x7FFFu + ((x >> 16) & 1u);          // round-to-nearest-even
  return (unsigned short)(x >> 16);
}

// ---------------------------------------------------------------------------
// Branch-free fragment loaders (rows/cols are pre-clamped outside the K loop;
// out-of-range output rows/cols are simply never stored by the epilogue).
//   A frag (16x32): lane m: elems = K[k0+hi*16 .. +16] of row m
//   B frag (32x16): lane n: elems = K[k0+hi*16 .. +16] of weight row n
// ---------------------------------------------------------------------------
__device__ __forceinline__ v16bf load_af(const unsigned short* __restrict__ p, int k0) {
  FragU f;
  f.u = *(const us16*)(p + k0);
  return f.v;
}

__device__ __forceinline__ v16bf load_bf(const unsigned short* __restrict__ p, int k0) {
  FragU f;
  f.u = *(const us16*)(p + k0);
  return f.v;
}

// Epilogue modes
#define EP_F32_BIAS 0   // Cf = acc + bias[col]
#define EP_TOK      1   // Cf = acc + bias[col] + pos[(row%196)*ldc + col]
#define EP_X2       2   // t = acc + bias + aux[row,col]; Cf=t; Cb=bf16(t)
#define EP_GELU     3   // Cb = bf16(gelu(acc + bias))
#define EP_MOE      4   // Cf[row,col] += aux[row*8+expert] * (acc + bias)
#define EP_BF16     5   // Cb = bf16(acc)
#define EP_PLAIN    6   // Cf = acc

// ---------------------------------------------------------------------------
// Batched WMMA GEMM: C[MxN] = A[MxK] * W[NxK]^T (+epilogue)
// 256 threads = 8 waves tiled 4(M) x 2(N); wave tile 32x64 (2x4 WMMA tiles).
// Block tile 128x128. Inner loop: 12 b128 loads -> 8 WMMAs, no branches.
// Batch offsets: z decomposed as zb = z / bmod, zh = z % bmod;
//   ptr += zb*off0 + zh*off1  (elements).
// ---------------------------------------------------------------------------
__global__ __launch_bounds__(256) void gemm_wmma_kernel(
    const unsigned short* __restrict__ A, int lda, long long az0, long long az1,
    const unsigned short* __restrict__ Bw, int ldb, long long bz0, long long bz1,
    float* __restrict__ Cf, unsigned short* __restrict__ Cb,
    int ldc, long long cz0, long long cz1, int bmod,
    const float* __restrict__ bias, const float* __restrict__ aux,
    int M, int N, int K, int mode, int expert) {
  const int z  = blockIdx.z;
  const int zb = z / bmod, zh = z % bmod;
  A  += (size_t)(zb * az0 + zh * az1);
  Bw += (size_t)(zb * bz0 + zh * bz1);
  const long long coff = zb * cz0 + zh * cz1;

  const int lane = threadIdx.x & 31;
  const int w    = threadIdx.x >> 5;
  const int wm   = w & 3;       // 0..3  (M bands of 32)
  const int wn   = w >> 2;      // 0..1  (N bands of 64)
  const int row0 = blockIdx.y * 128 + wm * 32;
  const int col0 = blockIdx.x * 128 + wn * 64;
  if (row0 >= M || col0 >= N) return;   // wave-uniform: EXEC stays full for WMMA

  const int m  = lane & 15;
  const int hi = lane >> 4;

  // Clamp addresses once; clamped (duplicate) rows/cols produce garbage only
  // in accumulator rows/cols that are never stored.
  int rA0 = row0 + m;        if (rA0 > M - 1) rA0 = M - 1;
  int rA1 = row0 + 16 + m;   if (rA1 > M - 1) rA1 = M - 1;
  const unsigned short* pa0 = A + (size_t)rA0 * lda + hi * 16;
  const unsigned short* pa1 = A + (size_t)rA1 * lda + hi * 16;

  const unsigned short* pb[4];
#pragma unroll
  for (int j = 0; j < 4; ++j) {
    int c = col0 + j * 16 + m;  if (c > N - 1) c = N - 1;
    pb[j] = Bw + (size_t)c * ldb + hi * 16;
  }

  v8f acc[2][4];
#pragma unroll
  for (int ti = 0; ti < 2; ++ti)
#pragma unroll
    for (int tj = 0; tj < 4; ++tj)
#pragma unroll
      for (int i = 0; i < 8; ++i) acc[ti][tj][i] = 0.f;

  for (int k0 = 0; k0 < K; k0 += 32) {
    // stream-ahead prefetch (lowers to global_prefetch_b8)
    __builtin_prefetch(pa0 + k0 + 64, 0, 0);
    __builtin_prefetch(pb[0] + k0 + 64, 0, 0);

    v16bf a0 = load_af(pa0, k0);
    v16bf a1 = load_af(pa1, k0);
    v16bf b0 = load_bf(pb[0], k0);
    v16bf b1 = load_bf(pb[1], k0);
    v16bf b2 = load_bf(pb[2], k0);
    v16bf b3 = load_bf(pb[3], k0);

    acc[0][0] = __builtin_amdgcn_wmma_f32_16x16x32_bf16(false, a0, false, b0, (short)0, acc[0][0], false, false);
    acc[0][1] = __builtin_amdgcn_wmma_f32_16x16x32_bf16(false, a0, false, b1, (short)0, acc[0][1], false, false);
    acc[0][2] = __builtin_amdgcn_wmma_f32_16x16x32_bf16(false, a0, false, b2, (short)0, acc[0][2], false, false);
    acc[0][3] = __builtin_amdgcn_wmma_f32_16x16x32_bf16(false, a0, false, b3, (short)0, acc[0][3], false, false);
    acc[1][0] = __builtin_amdgcn_wmma_f32_16x16x32_bf16(false, a1, false, b0, (short)0, acc[1][0], false, false);
    acc[1][1] = __builtin_amdgcn_wmma_f32_16x16x32_bf16(false, a1, false, b1, (short)0, acc[1][1], false, false);
    acc[1][2] = __builtin_amdgcn_wmma_f32_16x16x32_bf16(false, a1, false, b2, (short)0, acc[1][2], false, false);
    acc[1][3] = __builtin_amdgcn_wmma_f32_16x16x32_bf16(false, a1, false, b3, (short)0, acc[1][3], false, false);
  }

  // Epilogue. C/D layout: lane (hi,n), VGPR r -> C[hi*8+r][n] within 16x16 tile.
#pragma unroll
  for (int ti = 0; ti < 2; ++ti) {
#pragma unroll
    for (int tj = 0; tj < 4; ++tj) {
#pragma unroll
      for (int r = 0; r < 8; ++r) {
        const int row = row0 + ti * 16 + hi * 8 + r;
        const int col = col0 + tj * 16 + m;
        if (row >= M || col >= N) continue;
        float v = acc[ti][tj][r];
        const size_t ci = (size_t)coff + (size_t)row * ldc + col;
        if (mode == EP_PLAIN) {
          Cf[ci] = v;
        } else if (mode == EP_F32_BIAS) {
          Cf[ci] = v + bias[col];
        } else if (mode == EP_TOK) {
          Cf[ci] = v + bias[col] + aux[(size_t)(row % 196) * ldc + col];
        } else if (mode == EP_X2) {
          float t = v + bias[col] + aux[(size_t)row * ldc + col];
          Cf[ci] = t;
          Cb[ci] = f2bf(t);
        } else if (mode == EP_GELU) {
          float t = v + bias[col];
          t = 0.5f * t * (1.0f + erff(t * 0.70710678118654752f));
          Cb[ci] = f2bf(t);
        } else if (mode == EP_MOE) {
          float t = v + bias[col];
          Cf[ci] += aux[(size_t)row * 8 + expert] * t;
        } else { // EP_BF16
          Cb[ci] = f2bf(v);
        }
      }
    }
  }
}

// ---------------------------------------------------------------------------
// fp32 -> bf16 elementwise
// ---------------------------------------------------------------------------
__global__ __launch_bounds__(256) void cvt_bf16_kernel(const float* __restrict__ src,
                                                       unsigned short* __restrict__ dst, int n) {
  int i = blockIdx.x * 256 + threadIdx.x;
  if (i < n) dst[i] = f2bf(src[i]);
}

// ---------------------------------------------------------------------------
// im2col for the non-overlapping 16x16 patch conv: A0[t][c*256+p*16+q]
// ---------------------------------------------------------------------------
__global__ __launch_bounds__(256) void patchify_kernel(const float* __restrict__ x,
                                                       unsigned short* __restrict__ A0) {
  int idx = blockIdx.x * 256 + threadIdx.x;
  if (idx >= 3136 * 768) return;
  int kk = idx % 768, t = idx / 768;
  int c = kk >> 8, rem = kk & 255, p = rem >> 4, q = rem & 15;
  int b = t / 196, ij = t % 196, i = ij / 14, j = ij % 14;
  size_t src = (((size_t)(b * 3 + c)) * 224 + (i * 16 + p)) * 224 + (j * 16 + q);
  A0[idx] = f2bf(x[src]);
}

// ---------------------------------------------------------------------------
// LayerNorm over E=768, one 256-thread block per row. Optional f32/bf16 out.
// ---------------------------------------------------------------------------
__global__ __launch_bounds__(256) void ln_kernel(const float* __restrict__ X,
                                                 const float* __restrict__ g,
                                                 const float* __restrict__ b,
                                                 float* __restrict__ outF,
                                                 unsigned short* __restrict__ outB) {
  __shared__ float red[256];
  const int row = blockIdx.x, tid = threadIdx.x;
  const float* xr = X + (size_t)row * 768;
  float v0 = xr[tid], v1 = xr[tid + 256], v2 = xr[tid + 512];
  red[tid] = v0 + v1 + v2;
  __syncthreads();
  for (int o = 128; o > 0; o >>= 1) { if (tid < o) red[tid] += red[tid + o]; __syncthreads(); }
  const float mean = red[0] * (1.0f / 768.0f);
  __syncthreads();
  float d0 = v0 - mean, d1 = v1 - mean, d2 = v2 - mean;
  red[tid] = d0 * d0 + d1 * d1 + d2 * d2;
  __syncthreads();
  for (int o = 128; o > 0; o >>= 1) { if (tid < o) red[tid] += red[tid + o]; __syncthreads(); }
  const float inv = rsqrtf(red[0] * (1.0f / 768.0f) + 1e-5f);
  float o0 = d0 * inv * g[tid]       + b[tid];
  float o1 = d1 * inv * g[tid + 256] + b[tid + 256];
  float o2 = d2 * inv * g[tid + 512] + b[tid + 512];
  size_t base = (size_t)row * 768;
  if (outF) { outF[base + tid] = o0; outF[base + tid + 256] = o1; outF[base + tid + 512] = o2; }
  if (outB) { outB[base + tid] = f2bf(o0); outB[base + tid + 256] = f2bf(o1); outB[base + tid + 512] = f2bf(o2); }
}

// ---------------------------------------------------------------------------
// qkv [T,2304] f32 -> Q(bf16, x0.125) [BH,196,64], K(bf16) [BH,196,64],
// Vt(bf16) [BH,64,224] (columns 196..223 pre-zeroed via memset)
// ---------------------------------------------------------------------------
__global__ __launch_bounds__(256) void repack_qkv_kernel(const float* __restrict__ qkv,
                                                         unsigned short* __restrict__ Qb,
                                                         unsigned short* __restrict__ Kb,
                                                         unsigned short* __restrict__ Vt) {
  int idx = blockIdx.x * 256 + threadIdx.x;
  if (idx >= 16 * 12 * 196 * 64) return;
  int d = idx & 63, t = idx >> 6;
  int nn = t % 196, bh = t / 196;
  int h = bh % 12, b = bh / 12;
  size_t base = ((size_t)(b * 196 + nn)) * 2304 + h * 64 + d;
  float q = qkv[base] * 0.125f;          // 1/sqrt(64) folded into Q
  float k = qkv[base + 768];
  float v = qkv[base + 1536];
  size_t qi = (((size_t)bh) * 196 + nn) * 64 + d;
  Qb[qi] = f2bf(q);
  Kb[qi] = f2bf(k);
  Vt[(((size_t)bh) * 64 + d) * 224 + nn] = f2bf(v);
}

// ---------------------------------------------------------------------------
// Row softmax over 196 valid cols, write bf16 padded to 224 (zeros in pad).
// One 256-thread block per row (B*H*196 rows).
// ---------------------------------------------------------------------------
__global__ __launch_bounds__(256) void softmax_kernel(const float* __restrict__ S,
                                                      unsigned short* __restrict__ attb) {
  __shared__ float red[256];
  const int row = blockIdx.x, tid = threadIdx.x;
  float v = (tid < 196) ? S[(size_t)row * 196 + tid] : -3.4e38f;
  red[tid] = v;
  __syncthreads();
  for (int o = 128; o > 0; o >>= 1) { if (tid < o) red[tid] = fmaxf(red[tid], red[tid + o]); __syncthreads(); }
  const float mx = red[0];
  __syncthreads();
  float e = (tid < 196) ? __expf(v - mx) : 0.f;
  red[tid] = e;
  __syncthreads();
  for (int o = 128; o > 0; o >>= 1) { if (tid < o) red[tid] += red[tid + o]; __syncthreads(); }
  const float inv = 1.0f / red[0];
  if (tid < 224) attb[(size_t)row * 224 + tid] = f2bf((tid < 196) ? e * inv : 0.f);
}

// ---------------------------------------------------------------------------
// Router: logits = x2 @ router_w^T + router_b ; top-2 (ties -> lowest index);
// cnt[t][e] = 0.5 for chosen experts (the /K is folded here), else 0.
// ---------------------------------------------------------------------------
__global__ __launch_bounds__(256) void router_kernel(const float* __restrict__ x2,
                                                     const float* __restrict__ rw,
                                                     const float* __restrict__ rb,
                                                     float* __restrict__ cnt) {
  int t = blockIdx.x * 256 + threadIdx.x;
  if (t >= 3136) return;
  const float* xr = x2 + (size_t)t * 768;
  float lg[8];
#pragma unroll
  for (int e = 0; e < 8; ++e) {
    const float* w = rw + (size_t)e * 768;
    float s = rb[e];
    for (int i = 0; i < 768; ++i) s += xr[i] * w[i];
    lg[e] = s;
  }
  int i1 = 0;
#pragma unroll
  for (int e = 1; e < 8; ++e) if (lg[e] > lg[i1]) i1 = e;
  int i2 = (i1 == 0) ? 1 : 0;
#pragma unroll
  for (int e = 0; e < 8; ++e) if (e != i1 && lg[e] > lg[i2]) i2 = e;
#pragma unroll
  for (int e = 0; e < 8; ++e) cnt[(size_t)t * 8 + e] = (e == i1 || e == i2) ? 0.5f : 0.0f;
}

// ---------------------------------------------------------------------------
// Mean pool over N=196 tokens: pooled[b][e]
// ---------------------------------------------------------------------------
__global__ __launch_bounds__(256) void pool_kernel(const float* __restrict__ xn2,
                                                   float* __restrict__ pooled) {
  int idx = blockIdx.x * 256 + threadIdx.x;
  if (idx >= 16 * 768) return;
  int e = idx % 768, b = idx / 768;
  float s = 0.f;
  for (int n = 0; n < 196; ++n) s += xn2[((size_t)(b * 196 + n)) * 768 + e];
  pooled[idx] = s * (1.0f / 196.0f);
}

// ---------------------------------------------------------------------------
// Classifier head (tiny): out[b][c] = pooled[b] . head_w[c] + head_b[c]
// ---------------------------------------------------------------------------
__global__ __launch_bounds__(256) void head_kernel(const float* __restrict__ pooled,
                                                   const float* __restrict__ hw,
                                                   const float* __restrict__ hb,
                                                   float* __restrict__ out) {
  int idx = blockIdx.x * 256 + threadIdx.x;
  if (idx >= 16 * 1000) return;
  int c = idx % 1000, b = idx / 1000;
  const float* p = pooled + (size_t)b * 768;
  const float* w = hw + (size_t)c * 768;
  float s = hb[c];
  for (int i = 0; i < 768; ++i) s += p[i] * w[i];
  out[idx] = s;
}

// ---------------------------------------------------------------------------
// Host-side orchestration
// ---------------------------------------------------------------------------
extern "C" void kernel_launch(void* const* d_in, const int* in_sizes, int n_in,
                              void* d_out, int out_size, void* d_ws, size_t ws_size,
                              hipStream_t stream) {
  (void)in_sizes; (void)n_in; (void)out_size; (void)ws_size;
  const float* x          = (const float*)d_in[0];
  const float* patch_w    = (const float*)d_in[1];
  const float* patch_b    = (const float*)d_in[2];
  const float* pos_embed  = (const float*)d_in[3];
  const float* ln1_g      = (const float*)d_in[4];
  const float* ln1_b      = (const float*)d_in[5];
  const float* in_proj_w  = (const float*)d_in[6];
  const float* in_proj_b  = (const float*)d_in[7];
  const float* out_proj_w = (const float*)d_in[8];
  const float* out_proj_b = (const float*)d_in[9];
  const float* router_w   = (const float*)d_in[10];
  const float* router_b   = (const float*)d_in[11];
  const float* w1         = (const float*)d_in[12];
  const float* b1         = (const float*)d_in[13];
  const float* w2         = (const float*)d_in[14];
  const float* b2         = (const float*)d_in[15];
  const float* ln2_g      = (const float*)d_in[16];
  const float* ln2_b      = (const float*)d_in[17];
  const float* head_w     = (const float*)d_in[18];
  const float* head_b     = (const float*)d_in[19];

  const int T = 3136, E = 768, HID = 4608, BH = 192;

  char* ws = (char*)d_ws;
  size_t off = 0;
  auto alloc = [&](size_t bytes) -> void* {
    void* p = ws + off;
    off = (off + bytes + 255) & ~(size_t)255;
    return p;
  };
  unsigned short* Wp   = (unsigned short*)alloc((size_t)E * E * 2);
  unsigned short* Wq   = (unsigned short*)alloc((size_t)3 * E * E * 2);
  unsigned short* Wo   = (unsigned short*)alloc((size_t)E * E * 2);
  unsigned short* W1   = (unsigned short*)alloc((size_t)8 * HID * E * 2);
  unsigned short* W2   = (unsigned short*)alloc((size_t)8 * E * HID * 2);
  unsigned short* A0   = (unsigned short*)alloc((size_t)T * E * 2);
  float*          tok  = (float*)alloc((size_t)T * E * 4);
  unsigned short* xn   = (unsigned short*)alloc((size_t)T * E * 2);
  float*          qkv  = (float*)alloc((size_t)T * 3 * E * 4);
  unsigned short* Qb   = (unsigned short*)alloc((size_t)BH * 196 * 64 * 2);
  unsigned short* Kb   = (unsigned short*)alloc((size_t)BH * 196 * 64 * 2);
  unsigned short* Vt   = (unsigned short*)alloc((size_t)BH * 64 * 224 * 2);
  float*          S    = (float*)alloc((size_t)BH * 196 * 196 * 4);
  unsigned short* attb = (unsigned short*)alloc((size_t)BH * 196 * 224 * 2);
  unsigned short* o_bf = (unsigned short*)alloc((size_t)T * E * 2);
  float*          x2   = (float*)alloc((size_t)T * E * 4);
  unsigned short* x2b  = (unsigned short*)alloc((size_t)T * E * 2);
  float*          cnt  = (float*)alloc((size_t)T * 8 * 4);
  unsigned short* hbuf = (unsigned short*)alloc((size_t)T * HID * 2);
  float*          moe  = (float*)alloc((size_t)T * E * 4);
  float*          xn2  = (float*)alloc((size_t)T * E * 4);
  float*          pooled = (float*)alloc((size_t)16 * E * 4);

  auto cvt = [&](const float* s, unsigned short* d, int n) {
    cvt_bf16_kernel<<<(n + 255) / 256, 256, 0, stream>>>(s, d, n);
  };
  auto gemm = [&](const unsigned short* A, int lda, long long az0, long long az1,
                  const unsigned short* Bw, int ldb, long long bz0, long long bz1,
                  float* Cf, unsigned short* Cb, int ldc, long long cz0, long long cz1,
                  int bmod, const float* bias, const float* aux,
                  int M, int N, int K, int mode, int expert, int batches) {
    dim3 grid((N + 127) / 128, (M + 127) / 128, batches);
    gemm_wmma_kernel<<<grid, 256, 0, stream>>>(A, lda, az0, az1, Bw, ldb, bz0, bz1,
                                               Cf, Cb, ldc, cz0, cz1, bmod,
                                               bias, aux, M, N, K, mode, expert);
  };

  // Weight conversions fp32 -> bf16 ([out,in] row-major == WMMA B layout)
  cvt(patch_w,    Wp, E * E);
  cvt(in_proj_w,  Wq, 3 * E * E);
  cvt(out_proj_w, Wo, E * E);
  cvt(w1,         W1, 8 * HID * E);
  cvt(w2,         W2, 8 * E * HID);

  // Patch embed as GEMM: tok = A0 @ Wp^T + patch_b + pos_embed
  patchify_kernel<<<(T * E + 255) / 256, 256, 0, stream>>>(x, A0);
  gemm(A0, E, 0, 0, Wp, E, 0, 0, tok, nullptr, E, 0, 0, 1,
       patch_b, pos_embed, T, E, E, EP_TOK, 0, 1);

  // LN1 -> bf16, then QKV projection
  ln_kernel<<<T, 256, 0, stream>>>(tok, ln1_g, ln1_b, nullptr, xn);
  gemm(xn, E, 0, 0, Wq, E, 0, 0, qkv, nullptr, 3 * E, 0, 0, 1,
       in_proj_b, nullptr, T, 3 * E, E, EP_F32_BIAS, 0, 1);

  // Repack per-head Q (x0.125), K, V^T (K-dim zero-padded 196->224)
  hipMemsetAsync(Vt, 0, (size_t)BH * 64 * 224 * 2, stream);
  repack_qkv_kernel<<<(BH * 196 * 64 + 255) / 256, 256, 0, stream>>>(qkv, Qb, Kb, Vt);

  // scores[bh] = Q[bh] @ K[bh]^T   (K stored [196,64] row-major == B layout)
  gemm(Qb, 64, 196 * 64, 0, Kb, 64, 196 * 64, 0, S, nullptr, 196,
       (long long)196 * 196, 0, 1, nullptr, nullptr, 196, 196, 64, EP_PLAIN, 0, BH);

  softmax_kernel<<<BH * 196, 256, 0, stream>>>(S, attb);

  // O[bh] = att[bh] @ V[bh]; output scattered to o_bf[b*196+n][h*64+d]
  gemm(attb, 224, (long long)12 * 196 * 224, (long long)196 * 224,
       Vt,   224, (long long)12 * 64 * 224,  (long long)64 * 224,
       nullptr, o_bf, E, (long long)196 * E, 64, 12,
       nullptr, nullptr, 196, 64, 224, EP_BF16, 0, BH);

  // x2 = tok + o @ out_proj^T + b  (f32 + bf16 copies)
  gemm(o_bf, E, 0, 0, Wo, E, 0, 0, x2, x2b, E, 0, 0, 1,
       out_proj_b, tok, T, E, E, EP_X2, 0, 1);

  // Router top-2 (writes 0.5 per selected expert => /K folded in)
  router_kernel<<<(T + 255) / 256, 256, 0, stream>>>(x2, router_w, router_b, cnt);

  // Dense MoE: all 8 experts, count-weighted accumulate into moe
  hipMemsetAsync(moe, 0, (size_t)T * E * 4, stream);
  for (int e = 0; e < 8; ++e) {
    gemm(x2b, E, 0, 0, W1 + (size_t)e * HID * E, E, 0, 0,
         nullptr, hbuf, HID, 0, 0, 1, b1 + (size_t)e * HID, nullptr,
         T, HID, E, EP_GELU, 0, 1);
    gemm(hbuf, HID, 0, 0, W2 + (size_t)e * E * HID, HID, 0, 0,
         moe, nullptr, E, 0, 0, 1, b2 + (size_t)e * E, cnt,
         T, E, HID, EP_MOE, e, 1);
  }

  // LN2, mean pool, head
  ln_kernel<<<T, 256, 0, stream>>>(moe, ln2_g, ln2_b, xn2, nullptr);
  pool_kernel<<<(16 * E + 255) / 256, 256, 0, stream>>>(xn2, pooled);
  head_kernel<<<(16 * 1000 + 255) / 256, 256, 0, stream>>>(pooled, head_w, head_b, (float*)d_out);
}
